// BigBirdAttention_13511967113468
// MI455X (gfx1250) — compile-verified, matrix operand order checked by reading, not run on hardware
//
#include <hip/hip_runtime.h>

// ---------------------------------------------------------------------------
// Types for gfx1250 WMMA (wave32)
// ---------------------------------------------------------------------------
typedef __attribute__((ext_vector_type(16))) __bf16          v16bf;
typedef __attribute__((ext_vector_type(8)))  float           v8f;
typedef __attribute__((ext_vector_type(16))) unsigned short  u16x16;
typedef __attribute__((ext_vector_type(8)))  unsigned short  u16x8;
typedef __attribute__((ext_vector_type(4)))  int             v4i;

// ---------------------------------------------------------------------------
// Optional CDNA5 async global->LDS copies (guarded: falls back to sync path)
// Builtin signature (from hipcc diagnostic): void(v4i AS1*, v4i AS3*, imm, imm)
// ---------------------------------------------------------------------------
#if defined(__has_builtin)
#  if __has_builtin(__builtin_amdgcn_global_load_async_to_lds_b128) && \
      __has_builtin(__builtin_amdgcn_s_wait_asynccnt)
#    define USE_ASYNC_LDS 1
#  endif
#endif
#ifndef USE_ASYNC_LDS
#  define USE_ASYNC_LDS 0
#endif

#if USE_ASYNC_LDS
typedef __attribute__((address_space(1))) v4i g_v4i;
typedef __attribute__((address_space(3))) v4i l_v4i;
__device__ __forceinline__ void async_copy_b128(const void* g, void* l) {
    __builtin_amdgcn_global_load_async_to_lds_b128((g_v4i*)g, (l_v4i*)l, 0, 0);
}
__device__ __forceinline__ void async_wait0() {
    __builtin_amdgcn_s_wait_asynccnt(0);
}
#else
__device__ __forceinline__ void async_wait0() {}
#endif

__device__ __forceinline__ unsigned short f2bf(float f) {
    unsigned u = __builtin_bit_cast(unsigned, f);
    u += 0x7FFFu + ((u >> 16) & 1u);          // round-to-nearest-even
    return (unsigned short)(u >> 16);
}
__device__ __forceinline__ float bf2f(unsigned short s) {
    unsigned u = ((unsigned)s) << 16;
    return __builtin_bit_cast(float, u);
}

// Fragment gather for 16-bit WMMA operands.
// For a row holding 32 contiguous K-elements, lane (with half = lane>>4) needs
// elements {half*8 .. half*8+7} and {16+half*8 .. 16+half*8+7}.
__device__ __forceinline__ v16bf load_frag_row(const unsigned short* p, int half) {
    u16x8 lo = *(const u16x8*)(p + half * 8);
    u16x8 hi = *(const u16x8*)(p + 16 + half * 8);
    u16x16 t;
#pragma unroll
    for (int i = 0; i < 8; ++i) { t[i] = lo[i]; t[8 + i] = hi[i]; }
    return __builtin_bit_cast(v16bf, t);
}

__device__ __forceinline__ v8f wmma_bf16(v16bf a, v16bf b, v8f c) {
    return __builtin_amdgcn_wmma_f32_16x16x32_bf16(
        /*neg_a=*/false, a, /*neg_b=*/false, b,
        /*c_mod=*/(short)0, c, /*reuse_a=*/false, /*reuse_b=*/false);
}

// ---------------------------------------------------------------------------
// Problem constants
// ---------------------------------------------------------------------------
#define Bn   2
#define Tn   8193
#define Dn   1024
#define Hn   8
#define BSn  128
#define NBn  64               // (T-1)/BS
#define Mtot (Bn * Tn)        // 16386
#define INNER 512             // H*DK

// ---------------------------------------------------------------------------
// Conversion kernels
// ---------------------------------------------------------------------------
__global__ void k_f2bf(const float* __restrict__ src, unsigned short* __restrict__ dst, long n) {
    long i = (long)blockIdx.x * blockDim.x + threadIdx.x;
    if (i < n) dst[i] = f2bf(src[i]);
}

__global__ void k_pack_wqkv(const float* __restrict__ Wq, const float* __restrict__ Wk,
                            const float* __restrict__ Wv, unsigned short* __restrict__ dst) {
    long i = (long)blockIdx.x * blockDim.x + threadIdx.x;
    const long n = (long)Dn * 1536;
    if (i >= n) return;
    int kk = (int)(i / 1536), c = (int)(i % 1536);
    const float* s = (c < 512) ? Wq : (c < 1024 ? Wk : Wv);
    dst[i] = f2bf(s[(long)kk * 512 + (c & 511)]);
}

// ---------------------------------------------------------------------------
// bf16 WMMA GEMM: C(MxN) = A(MxK) * B(KxN)
// MODE 0: split output columns into q/k/v (bf16).  MODE 1: f32 out + bias.
// 256 threads = 8 waves; 128x128 C tile; wave tile 32x64.
// Double-buffered LDS: one barrier per 32-wide K chunk; A staged via async
// global->LDS when available, B staged transposed via registers.
// ---------------------------------------------------------------------------
template <int MODE>
__global__ void __launch_bounds__(256)
k_gemm(const unsigned short* __restrict__ A, const unsigned short* __restrict__ B,
       int M, int N, int K,
       unsigned short* __restrict__ oq, unsigned short* __restrict__ ok,
       unsigned short* __restrict__ ov,
       float* __restrict__ of, const float* __restrict__ bias) {
    __shared__ alignas(16) unsigned short Asb[2][128 * 32];   // [m][k]
    __shared__ alignas(16) unsigned short Bsb[2][128 * 32];   // transposed: [n][k]

    const int tid  = threadIdx.x;
    const int lane = tid & 31, wave = tid >> 5;
    const int half = (lane >> 4) & 1, lc = lane & 15;
    const int wm = wave & 3, wn = wave >> 2;
    const int nbase = blockIdx.x * 128;
    const int mbase = blockIdx.y * 128;

    v8f acc[2][4];
#pragma unroll
    for (int mi = 0; mi < 2; ++mi)
#pragma unroll
        for (int ni = 0; ni < 4; ++ni) acc[mi][ni] = (v8f){};

    const int ar = tid >> 1;            // A loader: row in tile 0..127
    const int ac = (tid & 1) * 16;      // col base 0 / 16
    const int bk = tid >> 3;            // B loader: k row 0..31
    const int bc = (tid & 7) * 16;      // n col base

    // stage chunk kc into buffer buf
    auto stage = [&](int buf, int kc) {
        {   // A (128x32) row-major
            long grow = (long)mbase + ar;
            unsigned short* d = &Asb[buf][ar * 32 + ac];
            if (grow < M) {
                const unsigned short* p = A + grow * (long)K + kc + ac;
#if USE_ASYNC_LDS
                async_copy_b128(p, d);
                async_copy_b128(p + 8, d + 8);
#else
                u16x8 x0 = *(const u16x8*)p;
                u16x8 x1 = *(const u16x8*)(p + 8);
                *(u16x8*)d       = x0;
                *(u16x8*)(d + 8) = x1;
#endif
                if (kc + 32 < K) __builtin_prefetch(p + 32, 0, 0);
            } else {
                u16x8 z = {};
                *(u16x8*)d       = z;
                *(u16x8*)(d + 8) = z;
            }
        }
        {   // B (32x128) transposed into [n][k]
            const unsigned short* p = B + (long)(kc + bk) * N + nbase + bc;
            u16x8 y0 = *(const u16x8*)p;
            u16x8 y1 = *(const u16x8*)(p + 8);
            if (kc + 32 < K) __builtin_prefetch(p + (long)32 * N, 0, 0);
#pragma unroll
            for (int e = 0; e < 8; ++e) {
                Bsb[buf][(bc + e) * 32 + bk]     = y0[e];
                Bsb[buf][(bc + 8 + e) * 32 + bk] = y1[e];
            }
        }
    };

    stage(0, 0);

    for (int kc = 0; kc < K; kc += 32) {
        const int buf = (kc >> 5) & 1;
        async_wait0();
        __syncthreads();

        // gather all fragments first (one dscnt wait covers them all)
        v16bf af[2], bfr[4];
#pragma unroll
        for (int mi = 0; mi < 2; ++mi)
            af[mi] = load_frag_row(&Asb[buf][(wm * 32 + mi * 16 + lc) * 32], half);
#pragma unroll
        for (int ni = 0; ni < 4; ++ni)
            bfr[ni] = load_frag_row(&Bsb[buf][(wn * 64 + ni * 16 + lc) * 32], half);

        // overlap next chunk staging with the WMMA burst
        if (kc + 32 < K) stage(buf ^ 1, kc + 32);

#pragma unroll
        for (int mi = 0; mi < 2; ++mi)
#pragma unroll
            for (int ni = 0; ni < 4; ++ni)
                acc[mi][ni] = wmma_bf16(af[mi], bfr[ni], acc[mi][ni]);
    }

    // Epilogue: C layout row = half*8 + r, col = lc (per 16x16 tile)
#pragma unroll
    for (int mi = 0; mi < 2; ++mi) {
#pragma unroll
        for (int ni = 0; ni < 4; ++ni) {
#pragma unroll
            for (int r = 0; r < 8; ++r) {
                long grow = (long)mbase + wm * 32 + mi * 16 + half * 8 + r;
                if (grow >= M) continue;
                int gcol = nbase + wn * 64 + ni * 16 + lc;
                float vv = acc[mi][ni][r];
                if (MODE == 0) {
                    int arr = gcol >> 9, lcol = gcol & 511;
                    unsigned short* o = (arr == 0) ? oq : (arr == 1 ? ok : ov);
                    o[grow * 512 + lcol] = f2bf(vv);
                } else {
                    of[grow * (long)N + gcol] = vv + bias[gcol];
                }
            }
        }
    }
}

// ---------------------------------------------------------------------------
// Global-token attention (row t=0 attends to all T keys)
// ---------------------------------------------------------------------------
__global__ void k_scores_g(const unsigned short* __restrict__ q,
                           const unsigned short* __restrict__ k,
                           float* __restrict__ sg) {
    long gid = (long)blockIdx.x * blockDim.x + threadIdx.x;
    const long TOT = (long)Bn * Hn * Tn;
    if (gid >= TOT) return;
    int t = (int)(gid % Tn);
    int h = (int)((gid / Tn) % Hn);
    int b = (int)(gid / ((long)Tn * Hn));
    const unsigned short* qp = q + ((long)b * Tn) * 512 + h * 64;
    const unsigned short* kp = k + ((long)b * Tn + t) * 512 + h * 64;
    float acc = 0.f;
#pragma unroll
    for (int d = 0; d < 64; ++d) acc += bf2f(qp[d]) * bf2f(kp[d]);
    sg[gid] = acc * 0.125f;
}

__global__ void __launch_bounds__(256)
k_global_attn(const float* __restrict__ sg, const unsigned short* __restrict__ v,
              unsigned short* __restrict__ attn) {
    __shared__ float red[256];
    const int tid = threadIdx.x;
    const int h = blockIdx.x & 7, b = blockIdx.x >> 3;
    const float* s = sg + ((long)b * Hn + h) * Tn;

    float lm = -3e38f;
    for (int t = tid; t < Tn; t += 256) lm = fmaxf(lm, s[t]);
    red[tid] = lm; __syncthreads();
    for (int st = 128; st > 0; st >>= 1) {
        if (tid < st) red[tid] = fmaxf(red[tid], red[tid + st]);
        __syncthreads();
    }
    float mx = red[0]; __syncthreads();

    float ls = 0.f;
    for (int t = tid; t < Tn; t += 256) ls += __expf(s[t] - mx);
    red[tid] = ls; __syncthreads();
    for (int st = 128; st > 0; st >>= 1) {
        if (tid < st) red[tid] += red[tid + st];
        __syncthreads();
    }
    float sum = red[0]; __syncthreads();

    const int d = tid & 63, g = tid >> 6;
    float acc = 0.f;
    for (int t = g; t < Tn; t += 4)
        acc += __expf(s[t] - mx) * bf2f(v[((long)b * Tn + t) * 512 + h * 64 + d]);
    red[tid] = acc; __syncthreads();
    if (tid < 64) {
        float tot = red[tid] + red[tid + 64] + red[tid + 128] + red[tid + 192];
        attn[((long)b * Tn) * 512 + h * 64 + tid] = f2bf(tot / sum);
    }
}

// ---------------------------------------------------------------------------
// Block-local BigBird attention (flash-style, WMMA bf16).
// 1 WG per (b,h,block): 8 waves x 16 query rows, 13 chunks of 32 keys.
// Keys: j=0 -> global kg; j in 1..384 -> tokens (n-1)*128+(j-1) (zero outside
// [0,8192) to match reference zero-padding); j>=385 masked to -inf.
// ---------------------------------------------------------------------------
__global__ void __launch_bounds__(256)
k_block_attn(const unsigned short* __restrict__ q, const unsigned short* __restrict__ k,
             const unsigned short* __restrict__ v, unsigned short* __restrict__ attn) {
    __shared__ alignas(16) unsigned short Kl[32 * 64];      // [key][kd]
    __shared__ alignas(16) unsigned short Vt[64 * 32];      // transposed [dv][key]
    __shared__ alignas(16) unsigned short Pl[8][16 * 32];   // per-wave P tile

    const int tid = threadIdx.x, lane = tid & 31, wave = tid >> 5;
    const int half = (lane >> 4) & 1, lc = lane & 15;
    const int bx = blockIdx.x;
    const int n = bx & 63, h = (bx >> 6) & 7, b = bx >> 9;
    const long base = (long)b * Tn;

    // Persistent Q fragments (16 queries x 64 dk = 2 A-frags)
    const int qtok = 1 + n * BSn + wave * 16 + lc;
    const unsigned short* qrow = q + (base + qtok) * 512 + h * 64;
    v16bf Qf0 = load_frag_row(qrow, half);
    v16bf Qf1 = load_frag_row(qrow + 32, half);

    v8f O0 = (v8f){}, O1 = (v8f){}, O2 = (v8f){}, O3 = (v8f){};
    float rmax[8], rsum[8];
#pragma unroll
    for (int r = 0; r < 8; ++r) { rmax[r] = -3e38f; rsum[r] = 0.f; }

    const int jr = tid >> 3;            // loader: key row in chunk 0..31
    const int cc = (tid & 7) * 8;       // loader: 8-elem col base

    for (int c = 0; c < 13; ++c) {
        __syncthreads();
        {   // cooperative stage of 32-key K/V chunk
            int j = c * 32 + jr;
            long row = -1;
            if (j == 0) row = base;
            else if (j <= 384) {
                int tok = (n - 1) * BSn + (j - 1);
                if (tok >= 0 && tok < NBn * BSn) row = base + 1 + tok;
            }
            if (row >= 0) {
                const unsigned short* kp = k + row * 512 + h * 64 + cc;
#if USE_ASYNC_LDS
                async_copy_b128(kp, &Kl[jr * 64 + cc]);   // straight copy: async path
#else
                *(u16x8*)&Kl[jr * 64 + cc] = *(const u16x8*)kp;
#endif
                u16x8 vv = *(const u16x8*)(v + row * 512 + h * 64 + cc);
#pragma unroll
                for (int e = 0; e < 8; ++e) Vt[(cc + e) * 32 + jr] = vv[e];
            } else {
                u16x8 z = {};
                *(u16x8*)&Kl[jr * 64 + cc] = z;
#pragma unroll
                for (int e = 0; e < 8; ++e) Vt[(cc + e) * 32 + jr] = 0;
            }
        }
        async_wait0();
        __syncthreads();

        // S = Q * K^T for two 16-key tiles; gather all frags, then WMMA burst
        v16bf Kf[4];
        Kf[0] = load_frag_row(&Kl[lc * 64 + 0], half);
        Kf[1] = load_frag_row(&Kl[(16 + lc) * 64 + 0], half);
        Kf[2] = load_frag_row(&Kl[lc * 64 + 32], half);
        Kf[3] = load_frag_row(&Kl[(16 + lc) * 64 + 32], half);
        v8f S0 = (v8f){}, S1 = (v8f){};
        S0 = wmma_bf16(Qf0, Kf[0], S0);
        S1 = wmma_bf16(Qf0, Kf[1], S1);
        S0 = wmma_bf16(Qf1, Kf[2], S0);
        S1 = wmma_bf16(Qf1, Kf[3], S1);

        const bool msk0 = (c * 32 + lc) >= 385;
        const bool msk1 = (c * 32 + 16 + lc) >= 385;
        unsigned short* Pw = &Pl[wave][0];
        float alpha[8];
#pragma unroll
        for (int r = 0; r < 8; ++r) {
            float s0 = msk0 ? -3e38f : S0[r] * 0.125f;
            float s1 = msk1 ? -3e38f : S1[r] * 0.125f;
            float cm = fmaxf(s0, s1);
#pragma unroll
            for (int m = 8; m >= 1; m >>= 1) cm = fmaxf(cm, __shfl_xor(cm, m, 32));
            float nm = fmaxf(rmax[r], cm);
            float p0 = __expf(s0 - nm);
            float p1 = __expf(s1 - nm);
            float ps = p0 + p1;
#pragma unroll
            for (int m = 8; m >= 1; m >>= 1) ps += __shfl_xor(ps, m, 32);
            alpha[r] = __expf(rmax[r] - nm);
            rsum[r] = rsum[r] * alpha[r] + ps;
            rmax[r] = nm;
            Pw[(half * 8 + r) * 32 + lc]      = f2bf(p0);
            Pw[(half * 8 + r) * 32 + 16 + lc] = f2bf(p1);
        }
#pragma unroll
        for (int r = 0; r < 8; ++r) {
            O0[r] *= alpha[r]; O1[r] *= alpha[r];
            O2[r] *= alpha[r]; O3[r] *= alpha[r];
        }

        // O += P * V: gather all V frags first, then WMMA burst
        v16bf Pf = load_frag_row(&Pw[lc * 32], half);
        v16bf Vf[4];
#pragma unroll
        for (int nt = 0; nt < 4; ++nt)
            Vf[nt] = load_frag_row(&Vt[(nt * 16 + lc) * 32], half);
        O0 = wmma_bf16(Pf, Vf[0], O0);
        O1 = wmma_bf16(Pf, Vf[1], O1);
        O2 = wmma_bf16(Pf, Vf[2], O2);
        O3 = wmma_bf16(Pf, Vf[3], O3);
    }

#pragma unroll
    for (int r = 0; r < 8; ++r) {
        float inv = 1.0f / rsum[r];
        int t = 1 + n * BSn + wave * 16 + half * 8 + r;
        unsigned short* orow = attn + (base + t) * 512 + h * 64;
        orow[0 * 16 + lc] = f2bf(O0[r] * inv);
        orow[1 * 16 + lc] = f2bf(O1[r] * inv);
        orow[2 * 16 + lc] = f2bf(O2[r] * inv);
        orow[3 * 16 + lc] = f2bf(O3[r] * inv);
    }
}

// ---------------------------------------------------------------------------
// Host-side launch
// ---------------------------------------------------------------------------
static inline size_t align256(size_t x) { return (x + 255) & ~(size_t)255; }

extern "C" void kernel_launch(void* const* d_in, const int* in_sizes, int n_in,
                              void* d_out, int out_size, void* d_ws, size_t ws_size,
                              hipStream_t stream) {
    (void)in_sizes; (void)n_in; (void)out_size; (void)ws_size;
    const float* x  = (const float*)d_in[0];
    const float* Wq = (const float*)d_in[1];
    const float* Wk = (const float*)d_in[2];
    const float* Wv = (const float*)d_in[3];
    const float* Wo = (const float*)d_in[4];
    const float* bo = (const float*)d_in[5];
    float* out = (float*)d_out;
    char* ws = (char*)d_ws;

    const long nx = (long)Mtot * Dn;          // 16,779,264
    const long nw = (long)Dn * 1536;          // packed Wqkv
    const long nwo = (long)INNER * Dn;        // 524,288
    const long nqkv = (long)Mtot * 512;       // q/k/v/attn each

    size_t off = 0;
    unsigned short* xb   = (unsigned short*)(ws + off); off = align256(off + nx * 2);
    unsigned short* wqkv = (unsigned short*)(ws + off); off = align256(off + nw * 2);
    unsigned short* wob  = (unsigned short*)(ws + off); off = align256(off + nwo * 2);
    unsigned short* qb   = (unsigned short*)(ws + off); off = align256(off + nqkv * 2);
    unsigned short* kb   = (unsigned short*)(ws + off); off = align256(off + nqkv * 2);
    unsigned short* vb   = (unsigned short*)(ws + off); off = align256(off + nqkv * 2);
    unsigned short* attn = (unsigned short*)(ws + off); off = align256(off + nqkv * 2);
    float* sg            = (float*)(ws + off);          off = align256(off + (long)Bn * Hn * Tn * 4);

    // 1. convert inputs to bf16
    k_f2bf<<<dim3((unsigned)((nx + 255) / 256)), 256, 0, stream>>>(x, xb, nx);
    k_pack_wqkv<<<dim3((unsigned)((nw + 255) / 256)), 256, 0, stream>>>(Wq, Wk, Wv, wqkv);
    k_f2bf<<<dim3((unsigned)((nwo + 255) / 256)), 256, 0, stream>>>(Wo, wob, nwo);

    // 2. QKV projection: (16386 x 1024) @ (1024 x 1536) -> q|k|v bf16
    k_gemm<0><<<dim3(1536 / 128, (Mtot + 127) / 128), 256, 0, stream>>>(
        xb, wqkv, Mtot, 1536, Dn, qb, kb, vb, nullptr, nullptr);

    // 3. global-token attention (t = 0)
    {
        long tot = (long)Bn * Hn * Tn;
        k_scores_g<<<dim3((unsigned)((tot + 255) / 256)), 256, 0, stream>>>(qb, kb, sg);
        k_global_attn<<<dim3(Bn * Hn), 256, 0, stream>>>(sg, vb, attn);
    }

    // 4. block-local attention: 2*8*64 = 1024 workgroups
    k_block_attn<<<dim3(Bn * Hn * NBn), 256, 0, stream>>>(qb, kb, vb, attn);

    // 5. output projection: (16386 x 512) @ (512 x 1024) + bo -> f32 out
    k_gemm<1><<<dim3(Dn / 128, (Mtot + 127) / 128), 256, 0, stream>>>(
        attn, wob, Mtot, Dn, INNER, nullptr, nullptr, nullptr, out, bo);
}
